// HadamardFeaturesModel_87608742903888
// MI455X (gfx1250) — matrix operand order repaired and checked by reading, not compiled, not run on hardware
//
#include <hip/hip_runtime.h>
#include <stdint.h>
#include <math.h>

typedef _Float16 v16h __attribute__((ext_vector_type(16)));
typedef _Float16 v8h  __attribute__((ext_vector_type(8)));
typedef float    v8f  __attribute__((ext_vector_type(8)));

#define NMOLS    256
#define NATOMS   32
#define RDIM     256
#define NPCAS    128
#define NSTACKS  32
#define NFEAT    4096
#define CHUNK_ATOMS 8
#define CHUNK_ROWS  (CHUNK_ATOMS * NSTACKS)   // 256 rows per chunk

// Padded LDS strides: multiples of 8 halfs (16B vector alignment) and odd multiples
// of 4 dwords mod 64 so 16 lanes hit 16 distinct banks.
#define SH_LD     136   // halfs, H matrix rows
#define SX_LD     136   // halfs, X staging rows
#define SREPH_LD  136   // halfs, rep f16 rows
#define SREP_LDF  264   // floats, rep f32 staging rows
#define RT_LD     264   // halfs, transposed reductor rows (256 K + pad)

// LDS regions (bytes, 16B aligned), with lifetime-based reuse:
//  sH   : Hadamard f16                      34816  (whole kernel)
//  r1   : sRepH f16 (8704) -> sSub f32      16384
//  r2   : sRedT f16 (67584) -> sX f16       69632
//  r3   : sRep f32 (33792) -> sAlpha f32    33792
#define SH_BYTES   (128 * SH_LD * 2)
#define R1_BYTES   (NATOMS * NPCAS * 4)
#define R2_BYTES   (CHUNK_ROWS * SX_LD * 2)
#define R3_BYTES   (NATOMS * SREP_LDF * 4)
#define SEI_BYTES  (NATOMS * 4)
#define LDS_TOTAL  (SH_BYTES + R1_BYTES + R2_BYTES + R3_BYTES + SEI_BYTES)

// ---- CDNA5 async global->LDS path (guarded; falls back to plain copies) ----
#if __has_builtin(__builtin_amdgcn_global_load_async_to_lds_b128) && \
    __has_builtin(__builtin_amdgcn_s_wait_asynccnt)
#define USE_ASYNC_LDS 1
#else
#define USE_ASYNC_LDS 0
#endif

typedef int i32x4 __attribute__((vector_size(16)));
typedef __attribute__((address_space(1))) i32x4* gptr128_t;   // global, prints as __device__
typedef __attribute__((address_space(3))) i32x4* lptr128_t;   // LDS, prints as __shared__

__device__ __forceinline__ void g2l_b128(const void* g, void* l) {
#if USE_ASYNC_LDS
    __builtin_amdgcn_global_load_async_to_lds_b128(
        (gptr128_t)(uintptr_t)g,
        (lptr128_t)(uint32_t)(uintptr_t)l,
        0, 0);
#else
    const float* gs = (const float*)g;
    float* ls = (float*)l;
    ls[0] = gs[0]; ls[1] = gs[1]; ls[2] = gs[2]; ls[3] = gs[3];
#endif
}
__device__ __forceinline__ void g2l_wait() {
#if USE_ASYNC_LDS
    __builtin_amdgcn_s_wait_asynccnt(0);
#endif
}

__global__ void zero_out_kernel(float* __restrict__ out, int n) {
    int i = blockIdx.x * blockDim.x + threadIdx.x;
    if (i < n) out[i] = 0.0f;
}

__global__ void __launch_bounds__(256)
hadamard_sorf_kernel(const float* __restrict__ rep,        // [256,32,256]
                     const int*   __restrict__ Z,          // [256,32]
                     const float* __restrict__ reductors,  // [4,256,128]
                     const float* __restrict__ Dmat,       // [4,2,32,128]
                     const float* __restrict__ bias,       // [4,4096]
                     const float* __restrict__ alpha,      // [4096]
                     float*       __restrict__ out)        // [256]
{
    extern __shared__ unsigned char smem[];
    _Float16* sH = (_Float16*)smem;
    unsigned char* r1 = smem + SH_BYTES;
    unsigned char* r2 = r1 + R1_BYTES;
    unsigned char* r3 = r2 + R2_BYTES;
    int* sEi = (int*)(r3 + R3_BYTES);

    _Float16* sRepH  = (_Float16*)r1;   // rep in f16, padded rows (stage A)
    float*    sSub   = (float*)r1;      // selected PCA output (stage B input)
    _Float16* sRedT  = (_Float16*)r2;   // reductors[e]^T in f16 (stage A)
    _Float16* sX     = (_Float16*)r2;   // SORF row staging (stage B)
    float*    sRep   = (float*)r3;      // rep f32 async landing zone
    float*    sAlpha = (float*)r3;      // alpha (stage C)

    const int m      = blockIdx.x;
    const int tid    = threadIdx.x;
    const int wave   = tid >> 5;
    const int lane   = tid & 31;
    const int laneLo = lane & 15;
    const int laneHi = lane >> 4;

    // ---- Phase 0: async-stage rep[m] (32x256 f32) into LDS; build H and element map.
    {
        const float* repm = rep + (size_t)m * NATOMS * RDIM;
        #pragma unroll
        for (int j = 0; j < 8; ++j) {                 // 2048 b128 transfers total
            int q  = tid + j * 256;
            int f0 = q * 4;                           // starting float index
            int row = f0 >> 8, col = f0 & 255;
            g2l_b128(repm + f0, sRep + row * SREP_LDF + col);
        }
    }
    if (tid < NATOMS) {
        int z = Z[m * NATOMS + tid];
        int e = -1;
        if (z == 1) e = 0; else if (z == 6) e = 1;
        else if (z == 7) e = 2; else if (z == 8) e = 3;
        sEi[tid] = e;
    }
    for (int i = tid; i < 128 * 128; i += 256) {      // H entries +/-1, exact in f16
        int r = i >> 7, c = i & 127;
        sH[r * SH_LD + c] = (_Float16)((__popc(r & c) & 1) ? -1.0f : 1.0f);
    }
    g2l_wait();
    __syncthreads();

    // Convert rep f32 -> padded f16 tile (done once; reused by all 4 element GEMMs).
    for (int i = tid; i < NATOMS * RDIM; i += 256) {
        int row = i >> 8, col = i & 255;
        sRepH[row * SREPH_LD + col] = (_Float16)sRep[row * SREP_LDF + col];
    }
    __syncthreads();                                   // sRep f32 dead after this

    // Kick off async staging of alpha into r3 (consumed in stage C).
    #pragma unroll
    for (int j = 0; j < 4; ++j) {
        int q = tid + j * 256;                         // 1024 b128 transfers
        g2l_b128(alpha + q * 4, sAlpha + q * 4);
    }

    // ---- Stage A: sub = rep @ reductors[e], element-selected per atom.
    // Wave w owns output columns [16w, 16w+16). Running register select -> one store.
    const int ncol = wave * 16 + laneLo;
    v8f sel0 = {0.f,0.f,0.f,0.f,0.f,0.f,0.f,0.f};
    v8f sel1 = sel0;
    for (int e = 0; e < 4; ++e) {
        __syncthreads();                               // prior sRedT readers done
        const float* re = reductors + (size_t)e * RDIM * NPCAS;
        for (int j = 0; j < 128; ++j) {                // coalesced read, transposed store
            int fl = tid + j * 256;
            int K = fl >> 7, n = fl & 127;
            sRedT[n * RT_LD + K] = (_Float16)re[fl];
        }
        __syncthreads();

        v8f acc0 = {0.f,0.f,0.f,0.f,0.f,0.f,0.f,0.f};
        v8f acc1 = acc0;
        for (int k = 0; k < 8; ++k) {
            const _Float16* bp = sRedT + ncol * RT_LD + k * 32 + laneHi * 8;
            v8h blo = *(const v8h*)bp;
            v8h bhi = *(const v8h*)(bp + 16);
            v16h bfrag = __builtin_shufflevector(blo, bhi, 0,1,2,3,4,5,6,7,8,9,10,11,12,13,14,15);

            const _Float16* a0p = sRepH + laneLo * SREPH_LD + k * 32 + laneHi * 8;
            v8h a0lo = *(const v8h*)a0p;
            v8h a0hi = *(const v8h*)(a0p + 16);
            v16h af0 = __builtin_shufflevector(a0lo, a0hi, 0,1,2,3,4,5,6,7,8,9,10,11,12,13,14,15);
            acc0 = __builtin_amdgcn_wmma_f32_16x16x32_f16(false, af0, false, bfrag,
                                                          (short)0, acc0, false, false);

            const _Float16* a1p = sRepH + (16 + laneLo) * SREPH_LD + k * 32 + laneHi * 8;
            v8h a1lo = *(const v8h*)a1p;
            v8h a1hi = *(const v8h*)(a1p + 16);
            v16h af1 = __builtin_shufflevector(a1lo, a1hi, 0,1,2,3,4,5,6,7,8,9,10,11,12,13,14,15);
            acc1 = __builtin_amdgcn_wmma_f32_16x16x32_f16(false, af1, false, bfrag,
                                                          (short)0, acc1, false, false);
        }
        // Branchless per-atom element select (v_cndmask, no exec manipulation).
        #pragma unroll
        for (int r = 0; r < 8; ++r) {
            int a0 = r + 8 * laneHi;
            int a1 = 16 + a0;
            sel0[r] = (sEi[a0] == e) ? acc0[r] : sel0[r];
            sel1[r] = (sEi[a1] == e) ? acc1[r] : sel1[r];
        }
    }
    __syncthreads();                                   // sRepH readers done; r1 becomes sSub
    #pragma unroll
    for (int r = 0; r < 8; ++r) {
        sSub[(r + 8 * laneHi) * NPCAS + ncol]      = sel0[r];
        sSub[(16 + r + 8 * laneHi) * NPCAS + ncol] = sel1[r];
    }
    g2l_wait();                                        // alpha landed
    __syncthreads();

    // ---- Stage B+C: SORF as two Hadamard WMMA GEMMs + fused cos/alpha reduction.
    const float scale = 0.0294627825f;                 // 1/(3*sqrt(128))
    const float fnorm = 0.0220970869f;                 // sqrt(2/4096)
    float local = 0.0f;

    for (int chunk = 0; chunk < NATOMS / CHUNK_ATOMS; ++chunk) {
        const int abase = chunk * CHUNK_ATOMS;

        // Build X[(a,s), p] = D1 .* sub[a] (f16), branchless masking.
        for (int i = tid; i < CHUNK_ROWS * NPCAS; i += 256) {
            int row = i >> 7, p = i & 127;
            int al = row >> 5, s = row & 31;
            int e = sEi[abase + al];
            int ec = (e < 0) ? 0 : e;
            float d1 = Dmat[(((size_t)ec * 2 + 0) * NSTACKS + s) * NPCAS + p];
            float sv = sSub[(abase + al) * NPCAS + p];
            float v = (e >= 0) ? d1 * sv : 0.0f;
            sX[row * SX_LD + p] = (_Float16)v;
        }
        __syncthreads();

        // Pass 1: Y = X @ H ; write back D2 .* Y in place (strip rows are private).
        for (int strip = wave; strip < CHUNK_ROWS / 16; strip += 8) {
            const int rowbase = strip * 16;
            v8f acc[8];
            #pragma unroll
            for (int i = 0; i < 8; ++i) acc[i] = (v8f){0.f,0.f,0.f,0.f,0.f,0.f,0.f,0.f};
            for (int k = 0; k < 4; ++k) {
                const _Float16* ap = sX + (rowbase + laneLo) * SX_LD + k * 32 + laneHi * 8;
                v8h alo = *(const v8h*)ap;
                v8h ahi = *(const v8h*)(ap + 16);
                v16h afrag = __builtin_shufflevector(alo, ahi, 0,1,2,3,4,5,6,7,8,9,10,11,12,13,14,15);
                #pragma unroll
                for (int nt = 0; nt < 8; ++nt) {
                    const _Float16* bp = sH + (nt * 16 + laneLo) * SH_LD + k * 32 + laneHi * 8;
                    v8h blo = *(const v8h*)bp;
                    v8h bhi = *(const v8h*)(bp + 16);
                    v16h bfrag = __builtin_shufflevector(blo, bhi, 0,1,2,3,4,5,6,7,8,9,10,11,12,13,14,15);
                    acc[nt] = __builtin_amdgcn_wmma_f32_16x16x32_f16(false, afrag, false, bfrag,
                                                                     (short)0, acc[nt], false, false);
                }
            }
            #pragma unroll
            for (int r = 0; r < 8; ++r) {
                int row = rowbase + r + 8 * laneHi;
                int al = row >> 5, s = row & 31;
                int e = sEi[abase + al];
                int ec = (e < 0) ? 0 : e;
                const float* d2row = Dmat + (((size_t)ec * 2 + 1) * NSTACKS + s) * NPCAS;
                #pragma unroll
                for (int nt = 0; nt < 8; ++nt) {
                    int col = nt * 16 + laneLo;
                    float v = (e >= 0) ? d2row[col] * acc[nt][r] : 0.0f;
                    sX[row * SX_LD + col] = (_Float16)v;
                }
            }
        }
        __syncthreads();

        // Pass 2: raw = (D2 .* Y) @ H ; fuse cos(scale*raw + bias) . alpha.
        for (int strip = wave; strip < CHUNK_ROWS / 16; strip += 8) {
            const int rowbase = strip * 16;
            v8f acc[8];
            #pragma unroll
            for (int i = 0; i < 8; ++i) acc[i] = (v8f){0.f,0.f,0.f,0.f,0.f,0.f,0.f,0.f};
            for (int k = 0; k < 4; ++k) {
                const _Float16* ap = sX + (rowbase + laneLo) * SX_LD + k * 32 + laneHi * 8;
                v8h alo = *(const v8h*)ap;
                v8h ahi = *(const v8h*)(ap + 16);
                v16h afrag = __builtin_shufflevector(alo, ahi, 0,1,2,3,4,5,6,7,8,9,10,11,12,13,14,15);
                #pragma unroll
                for (int nt = 0; nt < 8; ++nt) {
                    const _Float16* bp = sH + (nt * 16 + laneLo) * SH_LD + k * 32 + laneHi * 8;
                    v8h blo = *(const v8h*)bp;
                    v8h bhi = *(const v8h*)(bp + 16);
                    v16h bfrag = __builtin_shufflevector(blo, bhi, 0,1,2,3,4,5,6,7,8,9,10,11,12,13,14,15);
                    acc[nt] = __builtin_amdgcn_wmma_f32_16x16x32_f16(false, afrag, false, bfrag,
                                                                     (short)0, acc[nt], false, false);
                }
            }
            #pragma unroll
            for (int r = 0; r < 8; ++r) {
                int row = rowbase + r + 8 * laneHi;
                int al = row >> 5, s = row & 31;
                int e = sEi[abase + al];
                int ec = (e < 0) ? 0 : e;
                const float* brow = bias + (size_t)ec * NFEAT + s * 128;
                const float* arow = sAlpha + s * 128;
                #pragma unroll
                for (int nt = 0; nt < 8; ++nt) {
                    int p = nt * 16 + laneLo;
                    float c = __cosf(scale * acc[nt][r] + brow[p]);
                    local += (e >= 0) ? c * arow[p] : 0.0f;
                }
            }
        }
        __syncthreads();                               // protect sX before next chunk
    }

    atomicAdd(&out[m], fnorm * local);
}

extern "C" void kernel_launch(void* const* d_in, const int* in_sizes, int n_in,
                              void* d_out, int out_size, void* d_ws, size_t ws_size,
                              hipStream_t stream) {
    (void)in_sizes; (void)n_in; (void)d_ws; (void)ws_size;
    const float* rep       = (const float*)d_in[0];
    const int*   Z         = (const int*)  d_in[1];
    const float* reductors = (const float*)d_in[2];
    const float* Dmat      = (const float*)d_in[3];
    const float* bias      = (const float*)d_in[4];
    const float* alpha     = (const float*)d_in[5];
    float* out = (float*)d_out;

    zero_out_kernel<<<(out_size + 255) / 256, 256, 0, stream>>>(out, out_size);
    hadamard_sorf_kernel<<<NMOLS, 256, LDS_TOTAL, stream>>>(rep, Z, reductors, Dmat,
                                                            bias, alpha, out);
}
